// Head_3246995276528
// MI455X (gfx1250) — compile-verified, matrix operand order checked by reading, not run on hardware
//
#include <hip/hip_runtime.h>

// ---------------------------------------------------------------------------
// Causal single-head attention for MI455X (gfx1250, wave32, WMMA + TDM).
//   x:(8,2048,1024) f32, Wq/Wk/Wv:(1024,128) f32  ->  out:(8,2048,128) f32
// Kernel 1: QKV projection via v_wmma_f32_16x16x32_bf16 (fp32 accumulate),
//           Q pre-scaled by C^-0.5, V stored transposed [b][h][t].
// Kernel 2: flash-attention. Each block = 8 waves = 8 query tiles (same
//           batch). K/V 32-key tiles are DMA'd into double-buffered LDS by
//           the Tensor Data Mover (TENSORcnt-tracked), shared by all waves.
//           Scores & P*V via bf16 WMMA; P transposed through per-wave LDS.
// ---------------------------------------------------------------------------

typedef __attribute__((ext_vector_type(16))) __bf16 v16bf;
typedef __attribute__((ext_vector_type(8)))  __bf16 v8bf;
typedef __attribute__((ext_vector_type(8)))  float  v8f;
typedef __attribute__((ext_vector_type(4)))  unsigned int u32x4;
typedef __attribute__((ext_vector_type(8)))  int i32x8;
typedef __attribute__((ext_vector_type(4)))  int i32x4;

#define B_   8
#define T_   2048
#define C_   1024
#define H_   128

static __device__ __forceinline__ v8f wmma_bf16(v16bf a, v16bf b, v8f c) {
  return __builtin_amdgcn_wmma_f32_16x16x32_bf16(false, a, false, b,
                                                 (short)0, c, false, false);
}

// 16x32 bf16 fragment (A-layout / symmetric B-layout):
//   lanes 0-15: row = lane,    halves 0..7 = k 0..7,  halves 8..15 = k 16..23
//   lanes16-31: row = lane-16, halves 0..7 = k 8..15, halves 8..15 = k 24..31
static __device__ __forceinline__ v16bf load_frag(const __bf16* p, int ld) {
  const int lane = threadIdx.x & 31;
  const __bf16* q = p + (size_t)(lane & 15) * ld + ((lane >> 4) << 3);
  v8bf lo = *(const v8bf*)q;
  v8bf hi = *(const v8bf*)(q + 16);
  v16bf f;
#pragma unroll
  for (int i = 0; i < 8; ++i) { f[i] = lo[i]; f[i + 8] = hi[i]; }
  return f;
}

// --- Tensor Data Mover: 2D bf16 tile (tile_d1 rows x tile_d0 cols) -> LDS ---
// D# packing per CDNA5 ISA (group0: count/lds/global/
// type=2; group1: data_size=2B, tensor dims, tile dims, dim0 stride).
static __device__ __forceinline__ void tdm_load_2d(unsigned int lds_off,
                                                   const void* gptr,
                                                   unsigned int tensor_d0,
                                                   unsigned int tensor_d1,
                                                   unsigned int tile_d0,
                                                   unsigned int tile_d1,
                                                   unsigned int stride0) {
  const unsigned long long ga = (unsigned long long)(uintptr_t)gptr;
  u32x4 g0;
  g0[0] = 1u;                                              // count=1 (user D#)
  g0[1] = lds_off;                                         // lds_addr (bytes)
  g0[2] = (unsigned int)(ga & 0xffffffffu);                // global_addr[31:0]
  g0[3] = (unsigned int)((ga >> 32) & 0x1ffffffu) | (2u << 30); // [56:32]|type=2
  i32x8 g1;
  g1[0] = (int)(1u << 16);                                 // data_size=2 bytes
  g1[1] = (int)((tensor_d0 & 0xffffu) << 16);              // tensor_dim0 lo16
  g1[2] = (int)((tensor_d0 >> 16) | ((tensor_d1 & 0xffffu) << 16));
  g1[3] = (int)((tensor_d1 >> 16) | (tile_d0 << 16));      // tile_dim0
  g1[4] = (int)tile_d1;                                    // tile_dim1 (dim2=0)
  g1[5] = (int)stride0;                                    // dim0_stride[31:0]
  g1[6] = 0;
  g1[7] = 0;
  i32x4 gz = {0, 0, 0, 0};
#if defined(__clang_major__) && (__clang_major__ >= 23)
  i32x8 gz8 = {0, 0, 0, 0, 0, 0, 0, 0};
  __builtin_amdgcn_tensor_load_to_lds(g0, g1, gz, gz, gz8, 0);
#else
  __builtin_amdgcn_tensor_load_to_lds(g0, g1, gz, gz, 0);
#endif
}

// ---- Kernel 1: fused QKV projection --------------------------------------
__global__ void qkv_proj_kernel(const float* __restrict__ x,
                                const float* __restrict__ Wq,
                                const float* __restrict__ Wk,
                                const float* __restrict__ Wv,
                                __bf16* __restrict__ Q,
                                __bf16* __restrict__ K,
                                __bf16* __restrict__ Vt) {
  const int tile   = blockIdx.x;        // 16-row tile of flattened (B*T, C)
  const int which  = blockIdx.y;        // 0=Q 1=K 2=V
  const int wave   = threadIdx.x >> 5;  // 0..7 -> 16-col tile of H=128
  const int lane   = threadIdx.x & 31;
  const int colL   = lane & 15;
  const int halfHi = lane >> 4;
  const int rg     = tile * 16;
  const int col0   = wave * 16;
  const float* W = (which == 0) ? Wq : (which == 1) ? Wk : Wv;

  v8f acc = {0.f, 0.f, 0.f, 0.f, 0.f, 0.f, 0.f, 0.f};
  const float* xrow = x + (size_t)(rg + colL) * C_;

  for (int kc = 0; kc < C_; kc += 32) {
    const float* xp = xrow + kc + halfHi * 8;
    v16bf a;
#pragma unroll
    for (int i = 0; i < 8; ++i) { a[i] = (__bf16)xp[i]; a[i + 8] = (__bf16)xp[16 + i]; }
    const float* wp = W + (size_t)(kc + halfHi * 8) * H_ + col0 + colL;
    v16bf bm;
#pragma unroll
    for (int i = 0; i < 8; ++i) { bm[i] = (__bf16)wp[i * H_]; bm[i + 8] = (__bf16)wp[(16 + i) * H_]; }
    acc = wmma_bf16(a, bm, acc);
  }

  if (which == 0) {
#pragma unroll
    for (int r = 0; r < 8; ++r)
      Q[(size_t)(rg + r + 8 * halfHi) * H_ + col0 + colL] = (__bf16)(acc[r] * 0.03125f);
  } else if (which == 1) {
#pragma unroll
    for (int r = 0; r < 8; ++r)
      K[(size_t)(rg + r + 8 * halfHi) * H_ + col0 + colL] = (__bf16)acc[r];
  } else {
    const int b    = rg >> 11;
    const int tloc = rg & (T_ - 1);
#pragma unroll
    for (int r = 0; r < 8; ++r)
      Vt[(size_t)(b * H_ + col0 + colL) * T_ + tloc + r + 8 * halfHi] = (__bf16)acc[r];
  }
}

// ---- Kernel 2: flash attention with TDM-staged K/V -----------------------
// grid 128 (= 8 batches x 16 groups), block 256 (8 waves). Wave w owns query
// tile qt = group*8+w. All waves iterate to the block-max key block so the
// TDM double-buffer + barriers are block-uniform.
__global__ void attn_kernel(const __bf16* __restrict__ Q,
                            const __bf16* __restrict__ K,
                            const __bf16* __restrict__ Vt,
                            float* __restrict__ out) {
  __shared__ __bf16 kbuf[2][32][128];   // 16 KB: keys x head
  __shared__ __bf16 vbuf[2][128][32];   // 16 KB: head x keys
  __shared__ __bf16 plds[8][16][32];    //  8 KB: per-wave P transpose staging

  const int wave   = threadIdx.x >> 5;
  const int lane   = threadIdx.x & 31;
  const int colL   = lane & 15;
  const int halfHi = lane >> 4;
  const int b     = blockIdx.x >> 4;
  const int group = blockIdx.x & 15;
  const int qt    = group * 8 + wave;     // query tile 0..127
  const int q0    = qt * 16;
  const int nblk  = ((group * 8 + 7) >> 1) + 1;  // block-uniform trip count
  const int myblk = (qt >> 1) + 1;               // this wave's useful blocks

  const __bf16* Kb0 = K  + (size_t)b * T_ * H_;
  const __bf16* Vb0 = Vt + (size_t)b * H_ * T_;

  // Q fragments for the 16x128 query tile (pre-scaled by C^-0.5)
  const __bf16* Qb = Q + (size_t)(b * T_ + q0) * H_;
  v16bf qa[4];
#pragma unroll
  for (int c = 0; c < 4; ++c) qa[c] = load_frag(Qb + 32 * c, H_);

  v8f o[8];
#pragma unroll
  for (int h = 0; h < 8; ++h) o[h] = (v8f){0.f, 0.f, 0.f, 0.f, 0.f, 0.f, 0.f, 0.f};
  float mrow[8], lrow[8];
#pragma unroll
  for (int r = 0; r < 8; ++r) { mrow[r] = -1e30f; lrow[r] = 0.f; }

  // Preload key block 0 into buffer 0 (wave 0 drives the TDM).
  if (wave == 0) {
    tdm_load_2d((unsigned int)(uintptr_t)&kbuf[0][0][0], Kb0,
                H_, (unsigned int)(B_ * T_), H_, 32, H_);
    tdm_load_2d((unsigned int)(uintptr_t)&vbuf[0][0][0], Vb0,
                T_, (unsigned int)(B_ * H_), 32, H_, T_);
  }

  for (int j = 0; j < nblk; ++j) {
    const int key0 = j * 32;
    const int cur  = j & 1;
    if (wave == 0) {
      if (j + 1 < nblk) {             // pipeline: fetch next pair into other buf
        const int nk = key0 + 32;
        tdm_load_2d((unsigned int)(uintptr_t)&kbuf[cur ^ 1][0][0],
                    Kb0 + (size_t)nk * H_, H_, (unsigned int)(B_ * T_), H_, 32, H_);
        tdm_load_2d((unsigned int)(uintptr_t)&vbuf[cur ^ 1][0][0],
                    Vb0 + nk, T_, (unsigned int)(B_ * H_), 32, H_, T_);
        __builtin_amdgcn_s_wait_tensorcnt(2);   // current pair done, next in flight
      } else {
        __builtin_amdgcn_s_wait_tensorcnt(0);
      }
    }
    __syncthreads();                  // staged K/V visible to all waves

    if (j < myblk) {                  // wave-uniform guard (EXEC stays all-1s)
      // ---- scores: S = Q * K^T (two 16x16 tiles, chained over H=128) ----
      v8f s0 = {0.f, 0.f, 0.f, 0.f, 0.f, 0.f, 0.f, 0.f};
      v8f s1 = {0.f, 0.f, 0.f, 0.f, 0.f, 0.f, 0.f, 0.f};
#pragma unroll
      for (int c = 0; c < 4; ++c) {
        v16bf kb0 = load_frag(&kbuf[cur][0][0]  + 32 * c, H_);
        v16bf kb1 = load_frag(&kbuf[cur][16][0] + 32 * c, H_);
        s0 = wmma_bf16(qa[c], kb0, s0);
        s1 = wmma_bf16(qa[c], kb1, s1);
      }

      // ---- causal mask + online softmax (row = vgpr idx, col = lane) ----
#pragma unroll
      for (int r = 0; r < 8; ++r) {
        const int qrow = q0 + r + 8 * halfHi;
        float a  = s0[r];
        float bb = s1[r];
        if (key0 + colL > qrow)      a  = -1e30f;
        if (key0 + 16 + colL > qrow) bb = -1e30f;
        float mx = fmaxf(a, bb);
#pragma unroll
        for (int off = 1; off < 16; off <<= 1)
          mx = fmaxf(mx, __shfl_xor(mx, off, 32));
        const float mnew    = fmaxf(mrow[r], mx);
        const float rescale = __expf(mrow[r] - mnew);
        const float p0 = __expf(a - mnew);
        const float p1 = __expf(bb - mnew);
        float sum = p0 + p1;
#pragma unroll
        for (int off = 1; off < 16; off <<= 1)
          sum += __shfl_xor(sum, off, 32);
        lrow[r] = lrow[r] * rescale + sum;
        mrow[r] = mnew;
#pragma unroll
        for (int h = 0; h < 8; ++h) o[h][r] *= rescale;
        plds[wave][r + 8 * halfHi][colL]      = (__bf16)p0;
        plds[wave][r + 8 * halfHi][colL + 16] = (__bf16)p1;
      }

      // per-wave LDS fence before the cross-lane fragment reload
      asm volatile("s_wait_dscnt 0x0" ::: "memory");
      v16bf pa = load_frag(&plds[wave][0][0], 32);

      // ---- O += P * V ----
#pragma unroll
      for (int h = 0; h < 8; ++h) {
        v16bf vb = load_frag(&vbuf[cur][h * 16][0], 32);
        o[h] = wmma_bf16(pa, vb, o[h]);
      }
    }
    __syncthreads();                  // all reads done before buffer reuse
  }

  // ---- finalize: divide by row sums, store fp32 ----
  float* Ob = out + (size_t)(b * T_ + q0) * H_;
#pragma unroll
  for (int r = 0; r < 8; ++r) {
    const float inv = 1.0f / lrow[r];
    const int   tr  = r + 8 * halfHi;
#pragma unroll
    for (int h = 0; h < 8; ++h)
      Ob[(size_t)tr * H_ + h * 16 + colL] = o[h][r] * inv;
  }
}

// ---------------------------------------------------------------------------
extern "C" void kernel_launch(void* const* d_in, const int* in_sizes, int n_in,
                              void* d_out, int out_size, void* d_ws, size_t ws_size,
                              hipStream_t stream) {
  const float* x  = (const float*)d_in[0];
  const float* Wq = (const float*)d_in[1];
  const float* Wk = (const float*)d_in[2];
  const float* Wv = (const float*)d_in[3];

  const size_t mat_bytes = (size_t)B_ * T_ * H_ * 2;   // 4 MB each
  __bf16* Q  = (__bf16*)d_ws;
  __bf16* K  = (__bf16*)((char*)d_ws + mat_bytes);
  __bf16* Vt = (__bf16*)((char*)d_ws + 2 * mat_bytes);

  qkv_proj_kernel<<<dim3((B_ * T_) / 16, 3), 256, 0, stream>>>(x, Wq, Wk, Wv, Q, K, Vt);
  attn_kernel<<<dim3(B_ * (T_ / 16) / 8), 256, 0, stream>>>(Q, K, Vt, (float*)d_out);
}
